// QuotRemKVOPTAttention_15324443312232
// MI455X (gfx1250) — matrix operand order repaired and checked by reading, not compiled, not run on hardware
//
#include <hip/hip_runtime.h>

typedef __bf16 bf16;
typedef __attribute__((ext_vector_type(16))) __bf16 v16bf;
typedef __attribute__((ext_vector_type(8)))  float  v8f;

constexpr int S = 2048, E = 2048, NH = 32, HD = 64;

// ---------------- async global->LDS (CDNA5 GLOBAL_LOAD_ASYNC_TO_LDS_B128) ----------
// Builtin signature (from clang-22 diagnostics): (v4i as1*, v4i as3*, imm, imm)

typedef int v4i_vs __attribute__((vector_size(16)));
#if defined(__gfx1250__)
typedef __attribute__((address_space(1))) v4i_vs* gas_ptr;
typedef __attribute__((address_space(3))) v4i_vs* las_ptr;
#endif

__device__ __forceinline__ void async_copy16(void* lds, const void* g) {
#if defined(__gfx1250__) && __has_builtin(__builtin_amdgcn_global_load_async_to_lds_b128)
  __builtin_amdgcn_global_load_async_to_lds_b128((gas_ptr)g, (las_ptr)lds, 0, 0);
#else
  *(uint4*)lds = *(const uint4*)g;
#endif
}

__device__ __forceinline__ void async_join() {
#if defined(__gfx1250__) && __has_builtin(__builtin_amdgcn_s_wait_asynccnt)
  __builtin_amdgcn_s_wait_asynccnt(0);
#elif defined(__gfx1250__) && __has_builtin(__builtin_amdgcn_global_load_async_to_lds_b128)
  asm volatile("s_wait_asynccnt 0" ::: "memory");
#endif
}

// ---------------- WMMA fragment helpers (contiguous k = 16h + s order for A and B, wave32) ----------------

// A matrix 16x32 bf16: lanes 0-15 M=0..15 (K 0-15), lanes 16-31 same M
// (K 16-31). Caller pre-offsets lds to the tile's (mi,kk) corner.
__device__ __forceinline__ void load_a_frag(const bf16* lds, int ldl, int lane, v16bf& a) {
  const bf16* row = lds + (lane & 15) * ldl + ((lane >> 4) * 16);
  #pragma unroll
  for (int j = 0; j < 16; ++j) a[j] = row[j];
}

// B matrix 32x16 from an N-major (transposed, i.e. W[n][k]) LDS tile.
__device__ __forceinline__ void load_bT_frag(const bf16* lds, int ldl, int lane, v16bf& b) {
  const bf16* row = lds + (lane & 15) * ldl + (lane >> 4) * 16;
  #pragma unroll
  for (int j = 0; j < 16; ++j) b[j] = row[j];
}

// B matrix 32x16 from a K-major (natural B[k][n], e.g. V) LDS tile.
__device__ __forceinline__ void load_bN_frag(const bf16* lds, int ldl, int lane, v16bf& b) {
  const bf16* col = lds + (lane & 15) + (lane >> 4) * 16 * ldl;
  #pragma unroll
  for (int j = 0; j < 16; ++j) b[j] = col[j * ldl];
}

__device__ __forceinline__ v8f wmma_bf16(v16bf a, v16bf b, v8f c) {
  return __builtin_amdgcn_wmma_f32_16x16x32_bf16(false, a, false, b, (short)0, c,
                                                 false, false);
}

// ---------------- quotrem fake-quant (group of 16 = one C-tile row across lanes) ----

__device__ __forceinline__ float half16_max(float v) {
  v = fmaxf(v, __shfl_xor(v, 1, 32));
  v = fmaxf(v, __shfl_xor(v, 2, 32));
  v = fmaxf(v, __shfl_xor(v, 4, 32));
  v = fmaxf(v, __shfl_xor(v, 8, 32));
  return v;
}
__device__ __forceinline__ float half16_min(float v) {
  v = fminf(v, __shfl_xor(v, 1, 32));
  v = fminf(v, __shfl_xor(v, 2, 32));
  v = fminf(v, __shfl_xor(v, 4, 32));
  v = fminf(v, __shfl_xor(v, 8, 32));
  return v;
}

__device__ __forceinline__ float quotrem_quant(float x) {
  // Q_BITS=1 path, base_gs = r_gs = 16, R_BITS=3 -> r_q in [-4,3]
  float ma = fmaxf(half16_max(fabsf(x)), 1e-8f);
  float mx = half16_max(x);
  float mn = half16_min(x);
  float l2 = __log2f(ma);
  float bfl = exp2f(floorf(l2));
  float bce = exp2f(ceilf(l2));
  float base = (fabsf(ma - bfl) <= fabsf(bce - ma)) ? bfl : bce;
  base = fminf(fmaxf(base, 1.0f), 128.0f);
  float sf = (fabsf(mx) >= fabsf(mn)) ? 1.0f : -1.0f;
  float q  = (x * sf >= base * 0.5f) ? 1.0f : 0.0f;
  float qs = q * sf * base;
  float r  = x - qs;
  float sc = fmaxf(half16_max(fabsf(r)), 1e-8f) * (1.0f / 3.0f);
  float rq = fminf(fmaxf(rintf(r / sc), -4.0f), 3.0f);
  return qs + rq * sc;
}

// ---------------- fp32 -> bf16 convert ----------------

__global__ __launch_bounds__(256) void cvt_f32_bf16(const float* __restrict__ in,
                                                    bf16* __restrict__ out, int n) {
  int i = (blockIdx.x * 256 + threadIdx.x) * 4;
  if (i < n) {
    float4 f = *(const float4*)(in + i);
    out[i + 0] = (bf16)f.x;
    out[i + 1] = (bf16)f.y;
    out[i + 2] = (bf16)f.z;
    out[i + 3] = (bf16)f.w;
  }
}

// ---------------- projection GEMM: C = epi(A[M,K] @ W[N,K]^T + bias) ----------------
// Workgroup tile 64x128, 8 waves (2x4), each wave 32x32 = 2x2 WMMA tiles.
// MODE 0: (x+b)*scale -> bf16 (Q)   MODE 1: quant(x+b) -> bf16 (K,V)
// MODE 2: x+b -> f32 (output projection)
template <int MODE>
__global__ __launch_bounds__(256) void proj_gemm(
    const bf16* __restrict__ A, const bf16* __restrict__ W,
    const float* __restrict__ bias, float scale,
    bf16* __restrict__ outb, float* __restrict__ outf, int M, int N, int K) {
  __shared__ alignas(16) bf16 sA[64 * 32];
  __shared__ alignas(16) bf16 sB[128 * 32];
  const int tid = threadIdx.x;
  const int lane = tid & 31, wave = tid >> 5;
  const int m0 = blockIdx.x * 64, n0 = blockIdx.y * 128;
  const int mi = (wave >> 2) * 32, ni = (wave & 3) * 32;
  const int ar = tid >> 2, ac = (tid & 3) * 8;   // A tile: 64x32, 8 elem/thread
  const int br = tid >> 1, bc = (tid & 1) * 16;  // W tile: 128x32, 16 elem/thread
  v8f acc[2][2] = {};
  for (int k0 = 0; k0 < K; k0 += 32) {
    async_copy16(&sA[ar * 32 + ac], &A[(size_t)(m0 + ar) * K + k0 + ac]);
    async_copy16(&sB[br * 32 + bc], &W[(size_t)(n0 + br) * K + k0 + bc]);
    async_copy16(&sB[br * 32 + bc + 8], &W[(size_t)(n0 + br) * K + k0 + bc + 8]);
    async_join();
    __syncthreads();
    v16bf a[2], b[2];
    load_a_frag(sA + (mi + 0) * 32, 32, lane, a[0]);
    load_a_frag(sA + (mi + 16) * 32, 32, lane, a[1]);
    load_bT_frag(sB + (ni + 0) * 32, 32, lane, b[0]);
    load_bT_frag(sB + (ni + 16) * 32, 32, lane, b[1]);
    #pragma unroll
    for (int i = 0; i < 2; ++i)
      #pragma unroll
      for (int j = 0; j < 2; ++j)
        acc[i][j] = wmma_bf16(a[i], b[j], acc[i][j]);
    __syncthreads();
  }
  #pragma unroll
  for (int i = 0; i < 2; ++i) {
    #pragma unroll
    for (int j = 0; j < 2; ++j) {
      const int n  = n0 + ni + j * 16 + (lane & 15);
      const int mr = m0 + mi + i * 16 + (lane >> 4) * 8;
      const float bv = bias[n];
      #pragma unroll
      for (int r = 0; r < 8; ++r) {
        float x = acc[i][j][r] + bv;
        if (MODE == 0) {
          outb[(size_t)(mr + r) * N + n] = (bf16)(x * scale);
        } else if (MODE == 1) {
          outb[(size_t)(mr + r) * N + n] = (bf16)quotrem_quant(x);
        } else {
          outf[(size_t)(mr + r) * N + n] = x;
        }
      }
    }
  }
}

// ---------------- per-head causal scores: Sc[S,S] = Q_h @ K_h^T ----------------
// Workgroup tile 64x128, 8 waves (2x4), each wave 32x32; K = HD = 64 (2 steps).

__global__ __launch_bounds__(256) void scores_gemm(const bf16* __restrict__ Q,
                                                   const bf16* __restrict__ Kq,
                                                   float* __restrict__ Sc, int head) {
  const int tid = threadIdx.x;
  const int m0 = blockIdx.x * 64, n0 = blockIdx.y * 128;
  if (n0 > m0 + 63) {  // block entirely above the diagonal: all masked
    float* o = Sc + (size_t)m0 * S + n0;
    #pragma unroll
    for (int i = 0; i < 32; ++i) {
      int idx = tid + i * 256;
      o[(size_t)(idx >> 7) * S + (idx & 127)] = -__builtin_inff();
    }
    return;
  }
  __shared__ alignas(16) bf16 sA[64 * 64];
  __shared__ alignas(16) bf16 sB[128 * 64];
  const int lane = tid & 31, wave = tid >> 5;
  const int mi = (wave >> 2) * 32, ni = (wave & 3) * 32;
  {
    int r = tid >> 2, c = (tid & 3) * 16;
    const bf16* g = &Q[(size_t)(m0 + r) * E + head * HD + c];
    async_copy16(&sA[r * 64 + c], g);
    async_copy16(&sA[r * 64 + c + 8], g + 8);
  }
  {
    int r = tid >> 1, c = (tid & 1) * 32;
    const bf16* g = &Kq[(size_t)(n0 + r) * E + head * HD + c];
    async_copy16(&sB[r * 64 + c], g);
    async_copy16(&sB[r * 64 + c + 8], g + 8);
    async_copy16(&sB[r * 64 + c + 16], g + 16);
    async_copy16(&sB[r * 64 + c + 24], g + 24);
  }
  async_join();
  __syncthreads();
  v8f acc[2][2] = {};
  #pragma unroll
  for (int kk = 0; kk < HD; kk += 32) {
    v16bf a[2], b[2];
    load_a_frag(sA + (mi + 0) * 64 + kk, 64, lane, a[0]);
    load_a_frag(sA + (mi + 16) * 64 + kk, 64, lane, a[1]);
    load_bT_frag(sB + (ni + 0) * 64 + kk, 64, lane, b[0]);
    load_bT_frag(sB + (ni + 16) * 64 + kk, 64, lane, b[1]);
    #pragma unroll
    for (int i = 0; i < 2; ++i)
      #pragma unroll
      for (int j = 0; j < 2; ++j)
        acc[i][j] = wmma_bf16(a[i], b[j], acc[i][j]);
  }
  #pragma unroll
  for (int i = 0; i < 2; ++i) {
    #pragma unroll
    for (int j = 0; j < 2; ++j) {
      const int n  = n0 + ni + j * 16 + (lane & 15);
      const int mr = m0 + mi + i * 16 + (lane >> 4) * 8;
      #pragma unroll
      for (int r = 0; r < 8; ++r) {
        int m = mr + r;
        Sc[(size_t)m * S + n] = (n <= m) ? acc[i][j][r] : -__builtin_inff();
      }
    }
  }
}

// ---------------- row softmax: P[S,S] (bf16) ----------------

__global__ __launch_bounds__(256) void softmax_row(const float* __restrict__ Sc,
                                                   bf16* __restrict__ P) {
  __shared__ float red[256];
  const int row = blockIdx.x, tid = threadIdx.x;
  const float* src = Sc + (size_t)row * S;
  float v[8];
  float mx = -__builtin_inff();
  #pragma unroll
  for (int i = 0; i < 8; ++i) { v[i] = src[tid + i * 256]; mx = fmaxf(mx, v[i]); }
  red[tid] = mx;
  __syncthreads();
  for (int s = 128; s > 0; s >>= 1) {
    if (tid < s) red[tid] = fmaxf(red[tid], red[tid + s]);
    __syncthreads();
  }
  mx = red[0];
  __syncthreads();
  float sum = 0.f;
  #pragma unroll
  for (int i = 0; i < 8; ++i) { v[i] = __expf(v[i] - mx); sum += v[i]; }
  red[tid] = sum;
  __syncthreads();
  for (int s = 128; s > 0; s >>= 1) {
    if (tid < s) red[tid] += red[tid + s];
    __syncthreads();
  }
  const float inv = 1.0f / red[0];
  bf16* dst = P + (size_t)row * S;
  #pragma unroll
  for (int i = 0; i < 8; ++i) dst[tid + i * 256] = (bf16)(v[i] * inv);
}

// ---------------- per-head PV: Ctx_h[S,HD] = P @ V_h ----------------
// Workgroup tile 128x64, 8 waves (4x2), each wave 32x32; causal K truncation.

__global__ __launch_bounds__(256) void pv_gemm(const bf16* __restrict__ P,
                                               const bf16* __restrict__ Vq,
                                               bf16* __restrict__ Ctx, int head) {
  __shared__ alignas(16) bf16 sA[128 * 32];
  __shared__ alignas(16) bf16 sB[32 * 64];
  const int tid = threadIdx.x, lane = tid & 31, wave = tid >> 5;
  const int m0 = blockIdx.x * 128;
  const int mi = (wave >> 1) * 32, ni = (wave & 1) * 32;
  const int ar = tid >> 1, ac = (tid & 1) * 16;  // P tile: 128x32, 16/thread
  const int br = tid >> 3, bc = (tid & 7) * 8;   // V tile: 32x64, 8/thread
  v8f acc[2][2] = {};
  const int kmax = m0 + 128;  // causal: P[m,k]==0 for k>m
  for (int k0 = 0; k0 < kmax; k0 += 32) {
    const bf16* gp = &P[(size_t)(m0 + ar) * S + k0 + ac];
    async_copy16(&sA[ar * 32 + ac], gp);
    async_copy16(&sA[ar * 32 + ac + 8], gp + 8);
    async_copy16(&sB[br * 64 + bc], &Vq[(size_t)(k0 + br) * E + head * HD + bc]);
    async_join();
    __syncthreads();
    v16bf a[2], b[2];
    load_a_frag(sA + (mi + 0) * 32, 32, lane, a[0]);
    load_a_frag(sA + (mi + 16) * 32, 32, lane, a[1]);
    load_bN_frag(sB + ni + 0, 64, lane, b[0]);
    load_bN_frag(sB + ni + 16, 64, lane, b[1]);
    #pragma unroll
    for (int i = 0; i < 2; ++i)
      #pragma unroll
      for (int j = 0; j < 2; ++j)
        acc[i][j] = wmma_bf16(a[i], b[j], acc[i][j]);
    __syncthreads();
  }
  #pragma unroll
  for (int i = 0; i < 2; ++i) {
    #pragma unroll
    for (int j = 0; j < 2; ++j) {
      const int n  = ni + j * 16 + (lane & 15);
      const int mr = m0 + mi + i * 16 + (lane >> 4) * 8;
      #pragma unroll
      for (int r = 0; r < 8; ++r)
        Ctx[(size_t)(mr + r) * E + head * HD + n] = (bf16)acc[i][j][r];
    }
  }
}

// ---------------- host orchestration ----------------

extern "C" void kernel_launch(void* const* d_in, const int* in_sizes, int n_in,
                              void* d_out, int out_size, void* d_ws, size_t ws_size,
                              hipStream_t stream) {
  (void)in_sizes; (void)n_in; (void)out_size; (void)ws_size;
  const float* hs = (const float*)d_in[0];
  const float* Wq = (const float*)d_in[1];
  const float* bq = (const float*)d_in[2];
  const float* Wk = (const float*)d_in[3];
  const float* bk = (const float*)d_in[4];
  const float* Wv = (const float*)d_in[5];
  const float* bv = (const float*)d_in[6];
  const float* Wo = (const float*)d_in[7];
  const float* bo = (const float*)d_in[8];

  char* ws = (char*)d_ws;
  const size_t MB = 1ull << 20;
  bf16*  hsb = (bf16*)(ws + 0 * MB);
  bf16*  wqb = (bf16*)(ws + 8 * MB);
  bf16*  wkb = (bf16*)(ws + 16 * MB);
  bf16*  wvb = (bf16*)(ws + 24 * MB);
  bf16*  wob = (bf16*)(ws + 32 * MB);
  bf16*  qb  = (bf16*)(ws + 40 * MB);
  bf16*  kqb = (bf16*)(ws + 48 * MB);
  bf16*  vqb = (bf16*)(ws + 56 * MB);
  bf16*  ctx = (bf16*)(ws + 64 * MB);
  float* sc  = (float*)(ws + 72 * MB);  // 16 MB (one head's S x S scores)
  bf16*  pb  = (bf16*)(ws + 88 * MB);   // 8 MB  (one head's S x S probs)

  const int NE = S * E;
  const dim3 blk(256);
  const dim3 cg(NE / 1024);
  cvt_f32_bf16<<<cg, blk, 0, stream>>>(hs, hsb, NE);
  cvt_f32_bf16<<<cg, blk, 0, stream>>>(Wq, wqb, NE);
  cvt_f32_bf16<<<cg, blk, 0, stream>>>(Wk, wkb, NE);
  cvt_f32_bf16<<<cg, blk, 0, stream>>>(Wv, wvb, NE);
  cvt_f32_bf16<<<cg, blk, 0, stream>>>(Wo, wob, NE);

  const dim3 gp(S / 64, E / 128);
  proj_gemm<0><<<gp, blk, 0, stream>>>(hsb, wqb, bq, 0.125f, qb, nullptr, S, E, E);
  proj_gemm<1><<<gp, blk, 0, stream>>>(hsb, wkb, bk, 1.0f, kqb, nullptr, S, E, E);
  proj_gemm<1><<<gp, blk, 0, stream>>>(hsb, wvb, bv, 1.0f, vqb, nullptr, S, E, E);

  for (int h = 0; h < NH; ++h) {
    scores_gemm<<<dim3(S / 64, S / 128), blk, 0, stream>>>(qb, kqb, sc, h);
    softmax_row<<<dim3(S), blk, 0, stream>>>(sc, pb);
    pv_gemm<<<dim3(S / 128), blk, 0, stream>>>(pb, vqb, ctx, h);
  }

  proj_gemm<2><<<gp, blk, 0, stream>>>(ctx, wob, bo, 1.0f, nullptr, (float*)d_out,
                                       S, E, E);
}
